// InterFrameLambda_45311904973292
// MI455X (gfx1250) — matrix operand order repaired and checked by reading, not compiled, not run on hardware
//
#include <hip/hip_runtime.h>

// ---------------- constants ----------------
#define BD   8          // batch
#define NT   48         // spatial side
#define NPIX 2304       // NT*NT
#define CIN  128
#define HK   64         // heads*k
#define KD   16
#define VD   32
#define RLH  95         // 2*NT-1
#define ROWS (BD*NPIX)  // 18432
#define EMB_ELEMS (RLH*RLH*KD)   // 144400

typedef __bf16 v16bf __attribute__((ext_vector_type(16)));
typedef __bf16 v8bf  __attribute__((ext_vector_type(8)));
typedef float  v8f   __attribute__((ext_vector_type(8)));
typedef int    v4i_vs __attribute__((vector_size(16)));   // matches builtin's pointee type

#if defined(__has_builtin)
#  if __has_builtin(__builtin_amdgcn_global_load_async_to_lds_b128)
#    define HAVE_ASYNC_LDS 1
#  endif
#endif
#ifndef HAVE_ASYNC_LDS
#  define HAVE_ASYNC_LDS 0
#endif

// ---------------- WMMA operand builders ----------------
// A tile (16 rows x 32 K) from f32 row-major source, converting to bf16.
// Lane L: row = L%16 ; lanes<16 carry K {0..7,16..23}, lanes>=16 carry {8..15,24..31}.
__device__ __forceinline__ v16bf make_a_f32(const float* src, long row0, int ld, int kc) {
    int L = threadIdx.x & 31;
    int r = L & 15;
    int kb = (L < 16) ? 0 : 8;
    const float* p = src + (row0 + r) * (long)ld + kc + kb;
    v16bf a;
#pragma unroll
    for (int e = 0; e < 8; ++e) a[e] = (__bf16)p[e];
#pragma unroll
    for (int e = 0; e < 8; ++e) a[8 + e] = (__bf16)p[16 + e];
    return a;
}

// A tile from bf16 row-major source.
__device__ __forceinline__ v16bf make_a_bf16(const __bf16* src, long row0, int ld, int kc) {
    int L = threadIdx.x & 31;
    int r = L & 15;
    int kb = (L < 16) ? 0 : 8;
    const __bf16* p = src + (row0 + r) * (long)ld + kc + kb;
    v8bf lo = *(const v8bf*)(p);
    v8bf hi = *(const v8bf*)(p + 16);
    return __builtin_shufflevector(lo, hi, 0,1,2,3,4,5,6,7,8,9,10,11,12,13,14,15);
}

// B tile (32 K x 16 cols) from bf16 weights stored [out][k] row-major.
__device__ __forceinline__ v16bf make_b_bf16(const __bf16* w, int col0, int ld, int kc) {
    int L = threadIdx.x & 31;
    int c = col0 + (L & 15);
    int kk = kc + ((L < 16) ? 0 : 8);
    const __bf16* p = w + (long)c * ld + kk;
    v8bf lo = *(const v8bf*)(p);
    v8bf hi = *(const v8bf*)(p + 16);
    return __builtin_shufflevector(lo, hi, 0,1,2,3,4,5,6,7,8,9,10,11,12,13,14,15);
}

__device__ __forceinline__ v8f wmma_bf16(v16bf a, v16bf b, v8f c) {
    return __builtin_amdgcn_wmma_f32_16x16x32_bf16(false, a, false, b, (short)0, c, false, false);
}

// ---------------- prep: weights + emb table to bf16 ----------------
__global__ void prep_kernel(const float* Wq, const float* Wk, const float* Wv,
                            const float* Wcc, const float* Wmp, const float* emb,
                            __bf16* wqkv, __bf16* wccbf, __bf16* wmpbf, __bf16* embbf) {
    int tid = blockIdx.x * 256 + threadIdx.x;
    int stride = gridDim.x * 256;
    for (int i = tid; i < 112 * CIN; i += stride) {
        int o = i >> 7, k = i & 127;
        float v = (o < 64) ? Wq[i] : ((o < 80) ? Wk[(o - 64) * CIN + k] : Wv[(o - 80) * CIN + k]);
        wqkv[i] = (__bf16)v;
    }
    for (int i = tid; i < HK * CIN; i += stride) wccbf[i] = (__bf16)Wcc[i];
    for (int i = tid; i < CIN * HK; i += stride) wmpbf[i] = (__bf16)Wmp[i];
    for (int i = tid; i < EMB_ELEMS; i += stride) embbf[i] = (__bf16)emb[i];
}

// ---------------- proj: q/k/v = x @ W^T (WMMA), fused batch rotation + BN stats --------
__global__ __launch_bounds__(32) void proj_kernel(const float* x, const __bf16* wqkv,
                                                  float* q, float* kbuf, float* vbuf,
                                                  float* stats) {
    int rt = blockIdx.x, ct = blockIdx.y;
    int L = threadIdx.x;
    long row0 = (long)rt * 16;
    v8f acc = {};
#pragma unroll
    for (int kc = 0; kc < CIN; kc += 32) {
        v16bf a = make_a_f32(x, row0, CIN, kc);
        v16bf b = make_b_bf16(wqkv, ct * 16, CIN, kc);
        acc = wmma_bf16(a, b, acc);
    }
    int c = ct * 16 + (L & 15);
    int r8 = (L < 16) ? 0 : 8;
    int b0 = (int)(row0 / NPIX);        // whole tile in one batch (2304 % 16 == 0)
    int b2 = (b0 + 4) & 7;              // frame-reversed batch for k/v
    float s = 0.f, ss = 0.f;
#pragma unroll
    for (int r = 0; r < 8; ++r) {
        long grow = row0 + r8 + r;
        int nn = (int)(grow - (long)b0 * NPIX);
        float val = acc[r];
        if (c < 64) {
            q[grow * 64 + c] = val; s += val; ss += val * val;
        } else if (c < 80) {
            kbuf[((long)(b2 * KD + (c - 64))) * NPIX + nn] = val;
        } else {
            vbuf[((long)b2 * NPIX + nn) * VD + (c - 80)] = val; s += val; ss += val * val;
        }
    }
    if (ct != 4) {
        s  += __shfl_xor(s, 16, 32);
        ss += __shfl_xor(ss, 16, 32);
        if (L < 16) {
            int sidx = (c < 64) ? c : (128 + (c - 80));
            atomicAdd(&stats[sidx], s);
            atomicAdd(&stats[sidx + ((c < 64) ? 64 : 32)], ss);
        }
    }
}

// ---------------- BN finalize ----------------
__global__ void bn_finalize_kernel(const float* stats, const float* gq, const float* bq,
                                   const float* gv, const float* bv, float* normc) {
    int t = threadIdx.x;  // 96 threads
    const float inv_n = 1.0f / (float)ROWS;
    if (t < 64) {
        float mu = stats[t] * inv_n;
        float var = stats[64 + t] * inv_n - mu * mu;
        float sc = gq[t] * rsqrtf(var + 1e-5f);
        normc[t] = sc; normc[64 + t] = bq[t] - mu * sc;
    } else if (t < 96) {
        int cc = t - 64;
        float mu = stats[128 + cc] * inv_n;
        float var = stats[160 + cc] * inv_n - mu * mu;
        float sc = gv[cc] * rsqrtf(var + 1e-5f);
        normc[128 + cc] = sc; normc[160 + cc] = bv[cc] - mu * sc;
    }
}

// ---------------- softmax over m for each (b, kchan) ----------------
__global__ void softmax_kernel(const float* kbuf, float* ksm) {
    __shared__ float red[8];
    int t = threadIdx.x;
    const float* src = kbuf + (long)blockIdx.x * NPIX;
    float mx = -3.4e38f;
    for (int i = t; i < NPIX; i += 256) mx = fmaxf(mx, src[i]);
#pragma unroll
    for (int o = 16; o > 0; o >>= 1) mx = fmaxf(mx, __shfl_xor(mx, o, 32));
    if ((t & 31) == 0) red[t >> 5] = mx;
    __syncthreads();
    float bmx = red[0];
#pragma unroll
    for (int w = 1; w < 8; ++w) bmx = fmaxf(bmx, red[w]);
    __syncthreads();
    float s = 0.f;
    for (int i = t; i < NPIX; i += 256) s += __expf(src[i] - bmx);
#pragma unroll
    for (int o = 16; o > 0; o >>= 1) s += __shfl_xor(s, o, 32);
    if ((t & 31) == 0) red[t >> 5] = s;
    __syncthreads();
    float tot = 0.f;
#pragma unroll
    for (int w = 0; w < 8; ++w) tot += red[w];
    float inv = 1.0f / tot;
    for (int i = t; i < NPIX; i += 256)
        ksm[(long)blockIdx.x * NPIX + i] = __expf(src[i] - bmx) * inv;
}

// ---------------- pack normalized V into WMMA B-operand blocks (bf16) ----------------
__global__ void pack_v_kernel(const float* vbuf, const float* normc, __bf16* vbf) {
    int blk = blockIdx.x;
    int b = blk / 72, mc = blk % 72;
    int tt = threadIdx.x >> 5, L = threadIdx.x & 31;
    int v = tt * 16 + (L & 15);
    float sc = normc[128 + v], sh = normc[160 + v];
    int mbase = mc * 32 + ((L < 16) ? 0 : 8);
    __bf16* dst = vbf + (((long)(b * 72 + mc) * 2 + tt) * 512) + L * 16;
#pragma unroll
    for (int e = 0; e < 16; ++e) {
        float val = vbuf[((long)b * NPIX + mbase + e + ((e < 8) ? 0 : 8)) * VD + v] * sc + sh;
        dst[e] = (__bf16)val;
    }
}

// ---------------- pack QCE: rows 0-3 = normalized Q heads, rows 4-7 = cor_embed ------
__global__ void pack_qce_kernel(const float* q, const float* cor, const float* Wce,
                                const float* bce, const float* normc,
                                float* qce, float* cebuf) {
    long i = (long)blockIdx.x * 256 + threadIdx.x;  // over ROWS*128
    long rowi = i >> 7;
    int ch = (int)(i & 127);
    if (ch < 64) {
        qce[i] = q[rowi * 64 + ch] * normc[ch] + normc[64 + ch];
    } else {
        int o = ch - 64;
        float c0 = cor[rowi * 2], c1 = cor[rowi * 2 + 1];
        float val = c0 * Wce[o * 2] + c1 * Wce[o * 2 + 1] + bce[o];
        qce[i] = val;
        cebuf[rowi * 64 + o] = val;
    }
}

// ---------------- lamc[b][k][v] = sum_m softmaxK * Vn ----------------
__global__ void lamc_kernel(const float* ksm, const float* vbuf, const float* normc,
                            float* lamcb) {
    int b = blockIdx.x;
    int k = threadIdx.x >> 5, v = threadIdx.x & 31;
    float sc = normc[128 + v], sh = normc[160 + v];
    const float* kr = ksm + (long)(b * KD + k) * NPIX;
    const float* vr = vbuf + (long)b * NPIX * VD + v;
    float s = 0.f;
    for (int m = 0; m < NPIX; ++m) s += kr[m] * (vr[(long)m * VD] * sc + sh);
    lamcb[(b * KD + k) * VD + v] = s;
}

// ---------------- positional lambda + fused output contraction ----------------
// grid (144, 2), block 256. Each wave: 2 consecutive n  x  4 batches, so the scattered
// emb A-gather (depends on n only) is amortized over 8 WMMAs. Full bf16 rel-pos table
// (288.8 KB) lives in LDS; filled with the gfx1250 async global->LDS path.
__global__ __launch_bounds__(256, 1) void lambda_pos_kernel(const __bf16* embbf_g,
                                                            const __bf16* vbf,
                                                            const float* lamcb,
                                                            const float* qce,
                                                            float* CR, float* out_app) {
    extern __shared__ __bf16 semb[];
    int tid = threadIdx.x;
#if HAVE_ASYNC_LDS
    {   // 144400 bf16 = 18050 x b128 async copies, tracked by ASYNCcnt
        typedef __attribute__((address_space(1))) v4i_vs* g1p;
        typedef __attribute__((address_space(3))) v4i_vs* l3p;
        for (int i = tid; i < EMB_ELEMS / 8; i += 256) {
            g1p g = (g1p)(size_t)(const void*)(embbf_g + (long)i * 8);
            l3p l = (l3p)(unsigned int)(size_t)(void*)(semb + (long)i * 8);
            __builtin_amdgcn_global_load_async_to_lds_b128(g, l, 0, 0);
        }
#if __has_builtin(__builtin_amdgcn_s_wait_asynccnt)
        __builtin_amdgcn_s_wait_asynccnt(0);
#else
        asm volatile("s_wait_asynccnt 0" ::: "memory");
#endif
    }
#else
    {   // fallback: plain dword copy
        const unsigned int* src = (const unsigned int*)embbf_g;
        unsigned int* dst = (unsigned int*)semb;
        for (int i = tid; i < EMB_ELEMS / 2; i += 256) dst[i] = src[i];
    }
#endif
    __syncthreads();

    int wave = tid >> 5;
    int L = tid & 31;
    int kidx = L & 15;
    int mhalf = (L < 16) ? 0 : 8;
    int r8 = mhalf;
    int v0 = kidx;
    int b0 = blockIdx.y * 4;
    int nbase = blockIdx.x * 16 + wave * 2;

    for (int nn = 0; nn < 2; ++nn) {
        int n = nbase + nn;
        int iy = n / NT, ix = n - (n / NT) * NT;
        int cn = ((47 - iy) * RLH + (47 - ix)) * KD + kidx;  // lane-constant LDS offset
        v8f acc0[4] = {}, acc1[4] = {};
        for (int mc = 0; mc < 72; ++mc) {
            v16bf a;
            // run 1: m = mc*32 + mhalf + {0..7}
            int m = mc * 32 + mhalf;
            int ry = m / NT;
            int rx = m - ry * NT;
            int addr = ry * (RLH * KD) + rx * KD + cn;
#pragma unroll
            for (int e = 0; e < 8; ++e) {
                a[e] = semb[addr];
                rx += 1; addr += KD;
                int wrp = (rx >= NT);
                rx = wrp ? 0 : rx;
                addr += wrp ? (RLH - NT) * KD : 0;
            }
            // run 2: m = mc*32 + mhalf + 16 + {0..7}
            m = mc * 32 + mhalf + 16;
            ry = m / NT;
            rx = m - ry * NT;
            addr = ry * (RLH * KD) + rx * KD + cn;
#pragma unroll
            for (int e = 0; e < 8; ++e) {
                a[8 + e] = semb[addr];
                rx += 1; addr += KD;
                int wrp = (rx >= NT);
                rx = wrp ? 0 : rx;
                addr += wrp ? (RLH - NT) * KD : 0;
            }
            // same A multiplies V of 4 batches (batch stride 4608 v16bf blocks)
            const v16bf* vbB = ((const v16bf*)vbf) + (long)mc * 64 + L;
            __builtin_prefetch((const void*)(vbB + (long)b0 * 4608 + 64), 0, 1);
#pragma unroll
            for (int bb = 0; bb < 4; ++bb) {
                const v16bf* p = vbB + (long)(b0 + bb) * 4608;
                v16bf bm0 = p[0];
                v16bf bm1 = p[32];
                acc0[bb] = wmma_bf16(a, bm0, acc0[bb]);
                acc1[bb] = wmma_bf16(a, bm1, acc1[bb]);
            }
        }
#pragma unroll
        for (int bb = 0; bb < 4; ++bb) {
            int b = b0 + bb;
            v8f A0 = acc0[bb], A1 = acc1[bb];
            const float* lc = lamcb + b * (KD * VD);
#pragma unroll
            for (int r = 0; r < 8; ++r) {         // fuse lamc: L_total = lamc + lamp_n
                A0[r] += lc[(r8 + r) * VD + v0];
                A1[r] += lc[(r8 + r) * VD + 16 + v0];
            }
            const float* qrow = qce + ((long)b * NPIX + n) * 128;
            long obase = ((long)b * NPIX + n) * 128;
#pragma unroll
            for (int row = 0; row < 8; ++row) {   // out[row,v] = sum_k QCE[row,k]*L[k,v]
                const float* qk = qrow + row * 16 + r8;
                float s0 = 0.f, s1 = 0.f;
#pragma unroll
                for (int r = 0; r < 8; ++r) { s0 += qk[r] * A0[r]; s1 += qk[r] * A1[r]; }
                s0 += __shfl_xor(s0, 16, 32);     // combine k-halves
                s1 += __shfl_xor(s1, 16, 32);
                if (L < 16) {
                    long o = obase + (long)(row & 3) * 32;
                    if (row < 4) { out_app[o + v0] = s0; out_app[o + 16 + v0] = s1; }
                    else         { CR[o + v0] = s0;      CR[o + 16 + v0] = s1; }
                }
            }
        }
    }
}

// ---------------- cor_reverse: t2 = CR @ Wcc^T - cor_embed (WMMA) ----------------
__global__ __launch_bounds__(32) void correv_kernel(const float* CR, const __bf16* wccbf,
                                                    const float* cebuf, __bf16* t2bf) {
    int rt = blockIdx.x, ct = blockIdx.y;
    int L = threadIdx.x;
    long row0 = (long)rt * 16;
    v8f acc = {};
#pragma unroll
    for (int kc = 0; kc < CIN; kc += 32) {
        v16bf a = make_a_f32(CR, row0, CIN, kc);
        v16bf b = make_b_bf16(wccbf, ct * 16, CIN, kc);
        acc = wmma_bf16(a, b, acc);
    }
    int c = ct * 16 + (L & 15);
    int r8 = (L < 16) ? 0 : 8;
#pragma unroll
    for (int r = 0; r < 8; ++r) {
        long grow = row0 + r8 + r;
        t2bf[grow * 64 + c] = (__bf16)(acc[r] - cebuf[grow * 64 + c]);
    }
}

// ---------------- motion = t2 @ Wmp^T + bmp (WMMA) ----------------
__global__ __launch_bounds__(32) void motion_kernel(const __bf16* t2bf, const __bf16* wmpbf,
                                                    const float* bmp, float* out_mot) {
    int rt = blockIdx.x, ct = blockIdx.y;
    int L = threadIdx.x;
    long row0 = (long)rt * 16;
    v8f acc = {};
#pragma unroll
    for (int kc = 0; kc < 64; kc += 32) {
        v16bf a = make_a_bf16(t2bf, row0, 64, kc);
        v16bf b = make_b_bf16(wmpbf, ct * 16, 64, kc);
        acc = wmma_bf16(a, b, acc);
    }
    int c = ct * 16 + (L & 15);
    int r8 = (L < 16) ? 0 : 8;
    float bias = bmp[c];
#pragma unroll
    for (int r = 0; r < 8; ++r) {
        long grow = row0 + r8 + r;
        out_mot[grow * 128 + c] = acc[r] + bias;
    }
}

// ---------------- host launch ----------------
extern "C" void kernel_launch(void* const* d_in, const int* in_sizes, int n_in,
                              void* d_out, int out_size, void* d_ws, size_t ws_size,
                              hipStream_t stream) {
    (void)in_sizes; (void)n_in; (void)out_size; (void)ws_size;
    const float* x   = (const float*)d_in[0];
    const float* cor = (const float*)d_in[1];
    const float* Wq  = (const float*)d_in[2];
    const float* Wk  = (const float*)d_in[3];
    const float* Wv  = (const float*)d_in[4];
    const float* Wce = (const float*)d_in[5];
    const float* bce = (const float*)d_in[6];
    const float* Wcc = (const float*)d_in[7];
    const float* Wmp = (const float*)d_in[8];
    const float* bmp = (const float*)d_in[9];
    const float* gq  = (const float*)d_in[10];
    const float* bq  = (const float*)d_in[11];
    const float* gv  = (const float*)d_in[12];
    const float* bv  = (const float*)d_in[13];
    const float* emb = (const float*)d_in[14];

    char* wsb = (char*)d_ws;
    size_t off = 0;
    auto alloc = [&](size_t bytes) -> void* {
        void* p = wsb + off;
        off = (off + bytes + 255) & ~(size_t)255;
        return p;
    };
    float*  q      = (float*)alloc((size_t)ROWS * 64 * 4);
    float*  kbuf   = (float*)alloc((size_t)BD * KD * NPIX * 4);
    float*  vbuf   = (float*)alloc((size_t)ROWS * VD * 4);
    float*  stats  = (float*)alloc(192 * 4);
    float*  normc  = (float*)alloc(192 * 4);
    float*  ksm    = (float*)alloc((size_t)BD * KD * NPIX * 4);
    float*  lamcb  = (float*)alloc((size_t)BD * KD * VD * 4);
    __bf16* vbf    = (__bf16*)alloc((size_t)ROWS * VD * 2);
    __bf16* embbf  = (__bf16*)alloc((size_t)EMB_ELEMS * 2);
    float*  qce    = (float*)alloc((size_t)ROWS * 128 * 4);
    float*  CR     = (float*)alloc((size_t)ROWS * 128 * 4);
    float*  cebuf  = (float*)alloc((size_t)ROWS * 64 * 4);
    __bf16* t2bf   = (__bf16*)alloc((size_t)ROWS * 64 * 2);
    __bf16* wqkv   = (__bf16*)alloc((size_t)112 * CIN * 2);
    __bf16* wccbf  = (__bf16*)alloc((size_t)HK * CIN * 2);
    __bf16* wmpbf  = (__bf16*)alloc((size_t)CIN * HK * 2);

    float* out_app = (float*)d_out;
    float* out_mot = out_app + (size_t)ROWS * 128;

    (void)hipMemsetAsync(stats, 0, 192 * sizeof(float), stream);

    prep_kernel<<<576, 256, 0, stream>>>(Wq, Wk, Wv, Wcc, Wmp, emb, wqkv, wccbf, wmpbf, embbf);
    proj_kernel<<<dim3(ROWS / 16, 7), 32, 0, stream>>>(x, wqkv, q, kbuf, vbuf, stats);
    bn_finalize_kernel<<<1, 96, 0, stream>>>(stats, gq, bq, gv, bv, normc);
    softmax_kernel<<<BD * KD, 256, 0, stream>>>(kbuf, ksm);
    pack_v_kernel<<<BD * 72, 64, 0, stream>>>(vbuf, normc, vbf);
    pack_qce_kernel<<<(ROWS * 128) / 256, 256, 0, stream>>>(q, cor, Wce, bce, normc, qce, cebuf);
    lamc_kernel<<<BD, 512, 0, stream>>>(ksm, vbuf, normc, lamcb);

    size_t smem = (size_t)EMB_ELEMS * 2;  // 288,800 B of the 320 KB WGP LDS
    lambda_pos_kernel<<<dim3(NPIX / 16, 2), 256, smem, stream>>>(embbf, vbf, lamcb, qce, CR, out_app);

    correv_kernel<<<dim3(ROWS / 16, 4), 32, 0, stream>>>(CR, wccbf, cebuf, t2bf);
    motion_kernel<<<dim3(ROWS / 16, 8), 32, 0, stream>>>(t2bf, wmpbf, bmp, out_mot);
}
